// TransformerSeqLayer_15444702397194
// MI455X (gfx1250) — matrix operand order repaired and checked by reading, not compiled, not run on hardware
//
#include <hip/hip_runtime.h>

// ---------------------------------------------------------------------------
// Problem constants (from reference)
// ---------------------------------------------------------------------------
#define Bb     8
#define Mm     512
#define Ll     1024
#define Dd     512
#define NH     8
#define HDc    64          // Dd / NH
#define Ee     8
#define INNERc 2048
#define Tt     4096        // Bb * Mm
#define Sall   1536        // Ll + Mm
#define TOPK   2
#define NASSIGN 8192       // Tt * TOPK

typedef _Float16 f16;
typedef _Float16 v16h __attribute__((ext_vector_type(16)));
typedef _Float16 v8h  __attribute__((ext_vector_type(8)));
typedef _Float16 v4h  __attribute__((ext_vector_type(4)));
typedef float    v8f  __attribute__((ext_vector_type(8)));

typedef unsigned int u32x4 __attribute__((ext_vector_type(4)));
typedef int          i32x4 __attribute__((ext_vector_type(4)));
typedef int          i32x8 __attribute__((ext_vector_type(8)));

#if defined(__HIP_DEVICE_COMPILE__) && __has_builtin(__builtin_amdgcn_tensor_load_to_lds)
#define HAVE_TDM 1
#else
#define HAVE_TDM 0
#endif

union U16 { v16h v; v8h h[2]; };

__device__ __forceinline__ v16h make_frag(const f16* p0, const f16* p1) {
    U16 u;
    u.h[0] = *(const v8h*)p0;
    u.h[1] = *(const v8h*)p1;
    return u.v;
}

__device__ __forceinline__ v8f wmma_f16(v16h a, v16h b, v8f c) {
    // D = A(16x32 f16) * B(32x16 f16) + C(16x16 f32)   -> v_wmma_f32_16x16x32_f16
    return __builtin_amdgcn_wmma_f32_16x16x32_f16(false, a, false, b, (short)0, c, false, false);
}

__device__ __forceinline__ v8f zero_v8f() {
    v8f z;
    for (int i = 0; i < 8; ++i) z[i] = 0.f;
    return z;
}

// ---------------------------------------------------------------------------
// TDM: DMA a 2D f16 tile (tile_k contiguous elems x tile_n rows, row stride
// stride_elems) from global into LDS at lds_off, inserting 16B of padding
// after every 64B so a 32-f16 row lands with stride LDP=40 f16 (80B).
// D# layout per CDNA5 ISA.
// ---------------------------------------------------------------------------
#if HAVE_TDM
__device__ __forceinline__ void tdm_load_tile_f16(
    unsigned lds_off, const f16* gptr,
    unsigned tile_k, unsigned tile_n,
    unsigned tensor_k, unsigned tensor_n,
    unsigned long long stride_elems)
{
    unsigned long long ga = (unsigned long long)(size_t)gptr;
    u32x4 g0;
    g0[0] = 1u;                                              // count=1, user mode
    g0[1] = lds_off;                                         // lds_addr (bytes)
    g0[2] = (unsigned)(ga & 0xffffffffull);                  // global_addr[31:0]
    g0[3] = (unsigned)((ga >> 32) & 0x01ffffffull) | (2u << 30); // addr[56:32] | type=2
    i32x8 g1;
    // data_size=2B (code 1), pad_enable, pad_interval=16 DW (code 3), pad_amount=4 DW (code 3)
    g1[0] = (int)((1u << 16) | (1u << 20) | (3u << 22) | (3u << 25));
    g1[1] = (int)((tensor_k & 0xffffu) << 16);               // tensor_dim0[15:0]
    g1[2] = (int)(((tensor_k >> 16) & 0xffffu) | ((tensor_n & 0xffffu) << 16));
    g1[3] = (int)(((tensor_n >> 16) & 0xffffu) | ((tile_k & 0xffffu) << 16));
    g1[4] = (int)(tile_n & 0xffffu);                         // tile_dim1 (tile_dim2=0)
    g1[5] = (int)(stride_elems & 0xffffffffull);             // tensor_dim0_stride[31:0]
    g1[6] = (int)((stride_elems >> 32) & 0xffffull);         // stride[47:32]
    g1[7] = 0;
    i32x4 z4; z4[0] = 0; z4[1] = 0; z4[2] = 0; z4[3] = 0;
#if defined(__clang_major__) && (__clang_major__ >= 23)
    i32x8 z8;
    for (int i = 0; i < 8; ++i) z8[i] = 0;
    __builtin_amdgcn_tensor_load_to_lds(g0, g1, z4, z4, z8, 0);
#else
    __builtin_amdgcn_tensor_load_to_lds(g0, g1, z4, z4, 0);
#endif
}
#endif

// ---------------------------------------------------------------------------
// Weight prep: f32 [mat][K][N] -> f16 [mat][N][K]  (transpose + convert once)
// ---------------------------------------------------------------------------
__global__ __launch_bounds__(256) void k_w_t(const float* __restrict__ in,
                                             f16* __restrict__ out, int K, int N)
{
    size_t idx = (size_t)blockIdx.x * 256 + threadIdx.x;   // over mat*N*K, k fastest
    int k = (int)(idx % K);
    size_t rest = idx / K;
    int n = (int)(rest % N);
    size_t mat = rest / N;
    out[idx] = (f16)in[(mat * K + k) * N + n];
}

// ---------------------------------------------------------------------------
// Generic 128x128x32 WMMA GEMM; B is pre-transposed f16 [N][K] -> TDM-staged.
//  AMODE: 0 = f32 A direct rows; 1 = f32 A concat(h_cache,h) rows;
//         2 = f32 A expert-gathered rows (rowmap, bounded);
//         3 = f16 A (act) bucket rows (bounded) -> TDM-staged with HW zero-fill
//  OMODE: 0 = f16 out; 1 = f32 out; 2 = f16 out bounded (act);
//         3 = gate-weighted atomicAdd scatter into f32 (moe_out)
// ---------------------------------------------------------------------------
#define BMt 128
#define BNt 128
#define BKt 32
#define LDP 40   // padded f16 row stride (80B: 16B-aligned, conflict-free frags)

template <int AMODE, int OMODE, bool RELU>
__global__ __launch_bounds__(256) void k_gemm(
    const float* __restrict__ Af, const f16* __restrict__ Ah, const float* __restrict__ A2,
    const f16* __restrict__ Bt, const float* __restrict__ bias,
    float* __restrict__ Cf, f16* __restrict__ Ch,
    const int* __restrict__ rowmap, const float* __restrict__ rowgw,
    const int* __restrict__ meta,   // [0..7]=counts [8..15]=offsets [16..23]=cursors
    int Mr, int N, int K)
{
    const int e = blockIdx.z;
    int roff = 0, rcnt = Mr;
    if (AMODE >= 2) {
        roff = meta[8 + e];
        rcnt = meta[e];
        if ((int)blockIdx.y * BMt >= rcnt) return;
    }
    const f16* Bp = Bt + (size_t)e * K * N;
    const float* bp = bias ? bias + (size_t)e * N : nullptr;

    __shared__ alignas(16) f16 As[BMt * LDP];
    __shared__ alignas(16) f16 Bs[BNt * LDP];

    const int tid  = threadIdx.x;
    const int lane = tid & 31, wid = tid >> 5;
    const int hl   = lane >> 4, l15 = lane & 15;
    const int wm   = wid & 3,  wn  = wid >> 2;   // 4 x 2 wave grid; wave tile 32x64

    v8f acc[8];
    for (int i = 0; i < 8; ++i) acc[i] = zero_v8f();

    const int ktn = K >> 5;
    const int n0  = blockIdx.x * BNt;
    const int m0  = blockIdx.y * BMt;

    for (int kt = 0; kt < ktn; ++kt) {
        // ---- stage B tile: Bt[n0..n0+127][kt*32..+31] -> Bs (padded) ----
#if HAVE_TDM
        if (wid == 0) {
            tdm_load_tile_f16((unsigned)(size_t)(void*)Bs,
                              Bp + (size_t)n0 * K + kt * BKt,
                              BKt, BNt, (unsigned)(K - kt * BKt), (unsigned)(N - n0),
                              (unsigned long long)K);
        }
        if (AMODE == 3 && wid == 1) {
            tdm_load_tile_f16((unsigned)(size_t)(void*)As,
                              Ah + (size_t)(roff + m0) * K + kt * BKt,
                              BKt, BMt, (unsigned)(K - kt * BKt), (unsigned)(rcnt - m0),
                              (unsigned long long)K);
        }
#else
        for (int j = 0; j < 2; ++j) {
            int idx = tid + j * 256;                 // < 512
            int n = idx >> 2, ko = (idx & 3) << 3;
            v8h v = *(const v8h*)(Bp + (size_t)(n0 + n) * K + kt * BKt + ko);
            *(v8h*)&Bs[n * LDP + ko] = v;
        }
        if (AMODE == 3) {
            for (int j = 0; j < 2; ++j) {
                int idx = tid + j * 256;
                int rl = idx >> 2, ko = (idx & 3) << 3;
                int grl = m0 + rl;
                v8h v;
                if (grl < rcnt) {
                    v = *(const v8h*)(Ah + (size_t)(roff + grl) * K + kt * BKt + ko);
                } else {
                    for (int q = 0; q < 8; ++q) v[q] = (f16)0.f;
                }
                *(v8h*)&As[rl * LDP + ko] = v;
            }
        }
#endif
        // ---- stage A tile (f32 -> f16 convert) for f32 AMODEs ----
        if (AMODE != 3) {
            for (int j = 0; j < 4; ++j) {
                int idx = tid + j * 256;             // < 1024
                int rl = idx >> 3, kq = (idx & 7) << 2;
                const float* src = nullptr;
                if (AMODE == 0) {
                    src = Af + (size_t)(m0 + rl) * K;
                } else if (AMODE == 1) {
                    int gr = m0 + rl;
                    int bb = gr / Sall;
                    int s  = gr - bb * Sall;
                    src = (s < Ll) ? (A2 + ((size_t)bb * Ll + s) * K)
                                   : (Af + ((size_t)bb * Mm + (s - Ll)) * K);
                } else {  // AMODE == 2
                    int grl = m0 + rl;
                    if (grl < rcnt) src = Af + (size_t)rowmap[roff + grl] * K;
                }
                float4 v = src ? *(const float4*)(src + kt * BKt + kq)
                               : make_float4(0.f, 0.f, 0.f, 0.f);
                if (src && kt + 1 < ktn)
                    __builtin_prefetch(src + (kt + 1) * BKt + kq, 0, 0);
                v4h t;
                t[0] = (f16)v.x; t[1] = (f16)v.y; t[2] = (f16)v.z; t[3] = (f16)v.w;
                *(v4h*)&As[rl * LDP + kq] = t;
            }
        }
#if HAVE_TDM
        if (wid == 0 || (AMODE == 3 && wid == 1))
            __builtin_amdgcn_s_wait_tensorcnt(0);
#endif
        __syncthreads();

        // ---- compute: 2x4 WMMA tiles per wave ----
        v16h af[2], bf[4];
        for (int ms = 0; ms < 2; ++ms) {
            const f16* p = &As[(wm * 32 + ms * 16 + l15) * LDP];
            af[ms] = make_frag(p + hl * 16, p + hl * 16 + 8);
        }
        for (int ns = 0; ns < 4; ++ns) {
            const f16* p = &Bs[(wn * 64 + ns * 16 + l15) * LDP + hl * 16];
            bf[ns] = make_frag(p, p + 8);
        }
        for (int ms = 0; ms < 2; ++ms)
            for (int ns = 0; ns < 4; ++ns)
                acc[ms * 4 + ns] = wmma_f16(af[ms], bf[ns], acc[ms * 4 + ns]);
        __syncthreads();
    }

    // ---- epilogue ----
    for (int ms = 0; ms < 2; ++ms) {
        for (int r = 0; r < 8; ++r) {
            int rl  = wm * 32 + ms * 16 + r + 8 * hl;
            int grl = m0 + rl;
            int tok = 0; float gw = 0.f;
            if (OMODE == 3) {
                if (grl < rcnt) { tok = rowmap[roff + grl]; gw = rowgw[roff + grl]; }
            }
            for (int ns = 0; ns < 4; ++ns) {
                int col = n0 + wn * 64 + ns * 16 + l15;
                float v = acc[ms * 4 + ns][r];
                if (bp) v += bp[col];
                if (RELU) v = fmaxf(v, 0.f);
                if (OMODE == 0) {
                    Ch[(size_t)grl * N + col] = (f16)v;
                } else if (OMODE == 1) {
                    Cf[(size_t)grl * N + col] = v;
                } else if (OMODE == 2) {
                    if (grl < rcnt) Ch[(size_t)(roff + grl) * N + col] = (f16)v;
                } else {
                    if (grl < rcnt) atomicAdd(&Cf[(size_t)tok * N + col], gw * v);
                }
            }
        }
    }
}

// ---------------------------------------------------------------------------
// key_pe (HD x L) -> pe_t (L x HD) f16, so pos B-fragments are d-contiguous
// ---------------------------------------------------------------------------
__global__ __launch_bounds__(256) void k_pe_t(const float* __restrict__ key_pe,
                                              f16* __restrict__ pe_t)
{
    int idx = blockIdx.x * 256 + threadIdx.x;      // Ll*HDc = 65536
    int j = idx >> 6, d = idx & 63;
    pe_t[idx] = (f16)key_pe[(size_t)d * Ll + j];
}

// ---------------------------------------------------------------------------
// Flash attention over the unskew band: query m attends keys p = m .. m+L-1.
// Block = 4 waves, 64 query rows; wave w owns rows mb+16w..+15.
// p-chunks of 32 aligned to mb => pos term needs only j-tiles 16(tt-w) and
// the previous one (kept in per-wave 2-slot LDS, computed by WMMA).
// ---------------------------------------------------------------------------
__global__ __launch_bounds__(128) void k_attn(
    const f16* __restrict__ qh, const f16* __restrict__ kh, const f16* __restrict__ vh,
    const f16* __restrict__ pe_t, float* __restrict__ attn_out)
{
    const int bh = blockIdx.x;          // Bb*NH
    const int b  = bh >> 3, hh = bh & 7;
    const int mb = blockIdx.y * 64;
    const int tid  = threadIdx.x;
    const int lane = tid & 31, w = tid >> 5;
    const int hl   = lane >> 4, l15 = lane & 15;

    __shared__ alignas(16) f16   Vt[64 * LDP];         // V chunk transposed: [d][p]
    __shared__ alignas(16) float PosT[4][2][256];      // per-wave pos tiles (cur/prev)
    __shared__ alignas(16) f16   Pst[4][16 * LDP];     // per-wave P (16x32) staging

    // q fragments (resident): rows mb+16w+l15, K = HD = 64 -> 2 chunks
    const int mrow = mb + w * 16 + l15;
    const f16* qrow = qh + ((size_t)(b * Mm + mrow)) * Dd + hh * HDc;
    v16h qa[2];
    for (int kc = 0; kc < 2; ++kc)
        qa[kc] = make_frag(qrow + kc * 32 + hl * 16, qrow + kc * 32 + hl * 16 + 8);

    const f16* kbase = kh + ((size_t)b * Sall) * Dd + hh * HDc;
    const f16* vbase = vh + ((size_t)b * Sall) * Dd + hh * HDc;

    float mrun[8], lrun[8];
    v8f O[4];
    for (int r = 0; r < 8; ++r) { mrun[r] = -__builtin_inff(); lrun[r] = 0.f; }
    for (int ns = 0; ns < 4; ++ns) O[ns] = zero_v8f();

    const int NIT = 34;   // 68 p-subtiles cover p in [mb, mb+1087] (max 1535: in-bounds)
    for (int it = 0; it < NIT; ++it) {
        const int p0 = mb + it * 32;
        __syncthreads();   // prior PV reads of Vt complete

        // stage V chunk transposed (32 p x 64 d -> Vt[d][p])
        {
            int pr = tid >> 2, d0 = (tid & 3) << 4;
            const f16* vr = vbase + (size_t)(p0 + pr) * Dd + d0;
            v8h lo = *(const v8h*)vr;
            v8h hi = *(const v8h*)(vr + 8);
            for (int q = 0; q < 8; ++q) {
                Vt[(d0 + q) * LDP + pr]     = lo[q];
                Vt[(d0 + 8 + q) * LDP + pr] = hi[q];
            }
        }
        // per-wave relative-pos tiles (WMMA: qa x pe_t)
        for (int s = 0; s < 2; ++s) {
            int tt = it * 2 + s;
            int jb = (tt - w) * 16;
            if (jb >= 0 && jb <= Ll - 16) {
                v8f pc = zero_v8f();
                for (int kc = 0; kc < 2; ++kc) {
                    const f16* pr = pe_t + (size_t)(jb + l15) * HDc + kc * 32 + hl * 16;
                    v16h bf = make_frag(pr, pr + 8);
                    pc = wmma_f16(qa[kc], bf, pc);
                }
                float* dst = PosT[w][tt & 1];
                for (int r = 0; r < 8; ++r) dst[(r + 8 * hl) * 16 + l15] = pc[r];
            }
        }
        __syncthreads();   // Vt staged for all waves

        // scores for the two 16-col subtiles (K fragments straight from global)
        float sc[2][8];
        for (int s = 0; s < 2; ++s) {
            int tt = it * 2 + s;
            int pp = p0 + s * 16;
            v8f a = zero_v8f();
            for (int kc = 0; kc < 2; ++kc) {
                const f16* kr = kbase + (size_t)(pp + l15) * Dd + kc * 32 + hl * 16;
                v16h bf = make_frag(kr, kr + 8);
                a = wmma_f16(qa[kc], bf, a);
            }
            const float* cur = PosT[w][tt & 1];
            const float* prv = PosT[w][(tt + 1) & 1];
            for (int r = 0; r < 8; ++r) {
                int ml = r + 8 * hl;
                int j  = (tt - w) * 16 + l15 - ml;
                float pos = (l15 >= ml) ? cur[ml * 16 + (l15 - ml)]
                                        : prv[ml * 16 + (16 + l15 - ml)];
                float v = (a[r] + pos) * 0.125f;      // 1/sqrt(HD)
                sc[s][r] = (j >= 0 && j < Ll) ? v : -__builtin_inff();
            }
        }

        // online softmax (row stats reduced over the 16 lanes of each half)
        for (int r = 0; r < 8; ++r) {
            float rm = fmaxf(sc[0][r], sc[1][r]);
            for (int off = 1; off < 16; off <<= 1) rm = fmaxf(rm, __shfl_xor(rm, off, 32));
            float mn = fmaxf(mrun[r], rm);
            float alpha, pa, pb;
            if (mn == -__builtin_inff()) {
                alpha = 1.f; pa = 0.f; pb = 0.f;
            } else {
                alpha = __expf(mrun[r] - mn);
                pa = __expf(sc[0][r] - mn);
                pb = __expf(sc[1][r] - mn);
            }
            float rs = pa + pb;
            for (int off = 1; off < 16; off <<= 1) rs += __shfl_xor(rs, off, 32);
            lrun[r] = lrun[r] * alpha + rs;
            mrun[r] = mn;
            int ml = r + 8 * hl;
            Pst[w][ml * LDP + l15]      = (f16)pa;
            Pst[w][ml * LDP + 16 + l15] = (f16)pb;
            for (int ns = 0; ns < 4; ++ns) O[ns][r] *= alpha;
        }

        // O += P(16x32) @ V(32x64) -- P re-laid out C->A via per-wave LDS
        {
            const f16* pp = &Pst[w][l15 * LDP];
            v16h pa = make_frag(pp + hl * 16, pp + hl * 16 + 8);
            for (int ns = 0; ns < 4; ++ns) {
                const f16* vr = &Vt[(ns * 16 + l15) * LDP + hl * 16];
                v16h bf = make_frag(vr, vr + 8);
                O[ns] = wmma_f16(pa, bf, O[ns]);
            }
        }
    }

    // normalize + write (merge heads directly into [B*M, D] layout)
    for (int ns = 0; ns < 4; ++ns) {
        for (int r = 0; r < 8; ++r) {
            int m   = mb + w * 16 + r + 8 * hl;
            int col = hh * HDc + ns * 16 + l15;
            attn_out[((size_t)(b * Mm + m)) * Dd + col] = O[ns][r] / lrun[r];
        }
    }
}

// ---------------------------------------------------------------------------
// LayerNorm(Xa + Xb) -> out     (one block per row, 256 threads, D=512)
// ---------------------------------------------------------------------------
__global__ __launch_bounds__(256) void k_ln_resid(
    const float* __restrict__ Xa, const float* __restrict__ Xb,
    const float* __restrict__ g, const float* __restrict__ bb,
    float* __restrict__ out)
{
    int row = blockIdx.x, tid = threadIdx.x;
    __shared__ float red[8];
    size_t base = (size_t)row * Dd;
    float v0 = Xa[base + tid] + Xb[base + tid];
    float v1 = Xa[base + tid + 256] + Xb[base + tid + 256];
    float s = v0 + v1;
    for (int off = 16; off; off >>= 1) s += __shfl_xor(s, off, 32);
    if ((tid & 31) == 0) red[tid >> 5] = s;
    __syncthreads();
    if (tid == 0) { float t = 0; for (int i = 0; i < 8; ++i) t += red[i]; red[0] = t; }
    __syncthreads();
    float mu = red[0] * (1.f / Dd);
    __syncthreads();
    float d0 = v0 - mu, d1 = v1 - mu;
    float s2 = d0 * d0 + d1 * d1;
    for (int off = 16; off; off >>= 1) s2 += __shfl_xor(s2, off, 32);
    if ((tid & 31) == 0) red[tid >> 5] = s2;
    __syncthreads();
    if (tid == 0) { float t = 0; for (int i = 0; i < 8; ++i) t += red[i]; red[0] = t; }
    __syncthreads();
    float rstd = rsqrtf(red[0] * (1.f / Dd) + 1e-5f);
    out[base + tid]       = d0 * rstd * g[tid] + bb[tid];
    out[base + tid + 256] = d1 * rstd * g[tid + 256] + bb[tid + 256];
}

// ---------------------------------------------------------------------------
// Gate: scores = h1 @ Wg + bg (one wave per token), top-2, softmax weights,
// expert histogram via atomics.
// ---------------------------------------------------------------------------
__global__ __launch_bounds__(256) void k_gate(
    const float* __restrict__ h1, const float* __restrict__ Wg, const float* __restrict__ bg,
    int* __restrict__ topi, float* __restrict__ topw, int* __restrict__ meta)
{
    int lane = threadIdx.x & 31, w = threadIdx.x >> 5;
    int t = blockIdx.x * 8 + w;
    float s[Ee];
    for (int e = 0; e < Ee; ++e) s[e] = 0.f;
    for (int d = lane; d < Dd; d += 32) {
        float x = h1[(size_t)t * Dd + d];
        const float* wg = Wg + (size_t)d * Ee;
        for (int e = 0; e < Ee; ++e) s[e] += x * wg[e];
    }
    for (int e = 0; e < Ee; ++e)
        for (int off = 16; off; off >>= 1) s[e] += __shfl_xor(s[e], off, 32);
    if (lane == 0) {
        for (int e = 0; e < Ee; ++e) s[e] += bg[e];
        int i0 = 0; float v0 = s[0];
        for (int e = 1; e < Ee; ++e) if (s[e] > v0) { v0 = s[e]; i0 = e; }
        int i1 = -1; float v1 = -__builtin_inff();
        for (int e = 0; e < Ee; ++e) if (e != i0 && s[e] > v1) { v1 = s[e]; i1 = e; }
        float e1  = __expf(v1 - v0);
        float inv = 1.f / (1.f + e1);
        topi[2 * t] = i0;  topi[2 * t + 1] = i1;
        topw[2 * t] = inv; topw[2 * t + 1] = e1 * inv;
        atomicAdd(&meta[i0], 1);
        atomicAdd(&meta[i1], 1);
    }
}

__global__ void k_scan(int* __restrict__ meta)
{
    if (threadIdx.x == 0 && blockIdx.x == 0) {
        int acc = 0;
        for (int e = 0; e < Ee; ++e) {
            meta[8 + e]  = acc;   // offsets
            meta[16 + e] = acc;   // cursors
            acc += meta[e];
        }
    }
}

__global__ __launch_bounds__(256) void k_scatter(
    const int* __restrict__ topi, const float* __restrict__ topw,
    int* __restrict__ meta, int* __restrict__ rowmap, float* __restrict__ rowgw)
{
    int t = blockIdx.x * 256 + threadIdx.x;
    if (t >= Tt) return;
    for (int s = 0; s < TOPK; ++s) {
        int e = topi[2 * t + s];
        int pos = atomicAdd(&meta[16 + e], 1);
        rowmap[pos] = t;
        rowgw[pos]  = topw[2 * t + s];
    }
}

// ---------------------------------------------------------------------------
// mom_new = mu*momentum + gamma*moe ; h_out = LN(h1 - mom_new, ln2)
// d_out = [h_out (Tt*Dd) | mom_new (Tt*Dd)]
// ---------------------------------------------------------------------------
__global__ __launch_bounds__(256) void k_final(
    const float* __restrict__ h1, const float* __restrict__ moe, const float* __restrict__ momi,
    const float* __restrict__ g, const float* __restrict__ bb, float* __restrict__ out)
{
    int row = blockIdx.x, tid = threadIdx.x;
    __shared__ float red[8];
    size_t base = (size_t)row * Dd;
    float mn0 = 0.7f * momi[base + tid]       + 1.0f * moe[base + tid];
    float mn1 = 0.7f * momi[base + tid + 256] + 1.0f * moe[base + tid + 256];
    float v0 = h1[base + tid] - mn0;
    float v1 = h1[base + tid + 256] - mn1;
    float s = v0 + v1;
    for (int off = 16; off; off >>= 1) s += __shfl_xor(s, off, 32);
    if ((tid & 31) == 0) red[tid >> 5] = s;
    __syncthreads();
    if (tid == 0) { float t = 0; for (int i = 0; i < 8; ++i) t += red[i]; red[0] = t; }
    __syncthreads();
    float mu = red[0] * (1.f / Dd);
    __syncthreads();
    float d0 = v0 - mu, d1 = v1 - mu;
    float s2 = d0 * d0 + d1 * d1;
    for (int off = 16; off; off >>= 1) s2 += __shfl_xor(s2, off, 32);
    if ((tid & 31) == 0) red[tid >> 5] = s2;
    __syncthreads();
    if (tid == 0) { float t = 0; for (int i = 0; i < 8; ++i) t += red[i]; red[0] = t; }
    __syncthreads();
    float rstd = rsqrtf(red[0] * (1.f / Dd) + 1e-5f);
    out[base + tid]       = d0 * rstd * g[tid] + bb[tid];
    out[base + tid + 256] = d1 * rstd * g[tid + 256] + bb[tid + 256];
    size_t mo = (size_t)Tt * Dd;
    out[mo + base + tid]       = mn0;
    out[mo + base + tid + 256] = mn1;
}

// ---------------------------------------------------------------------------
// Host orchestration
// ---------------------------------------------------------------------------
extern "C" void kernel_launch(void* const* d_in, const int* in_sizes, int n_in,
                              void* d_out, int out_size, void* d_ws, size_t ws_size,
                              hipStream_t stream)
{
    (void)in_sizes; (void)n_in; (void)out_size; (void)ws_size;
    const float* h      = (const float*)d_in[0];
    const float* hcache = (const float*)d_in[1];
    const float* key_pe = (const float*)d_in[2];
    const float* momi   = (const float*)d_in[3];
    const float* Wq     = (const float*)d_in[4];
    const float* Wk     = (const float*)d_in[5];
    const float* Wv     = (const float*)d_in[6];
    const float* Wo     = (const float*)d_in[7];
    const float* ln1g   = (const float*)d_in[8];
    const float* ln1b   = (const float*)d_in[9];
    const float* Wg     = (const float*)d_in[10];
    const float* bg     = (const float*)d_in[11];
    const float* W1     = (const float*)d_in[12];
    const float* b1     = (const float*)d_in[13];
    const float* W2     = (const float*)d_in[14];
    const float* b2     = (const float*)d_in[15];
    const float* ln2g   = (const float*)d_in[16];
    const float* ln2b   = (const float*)d_in[17];
    float* out = (float*)d_out;

    char* ws = (char*)d_ws;
    auto carve = [&](size_t bytes) -> char* {
        char* p = ws;
        ws += (bytes + 255) & ~(size_t)255;
        return p;
    };
    f16*   qh     = (f16*)  carve((size_t)Tt * Dd * sizeof(f16));
    f16*   kh     = (f16*)  carve((size_t)Bb * Sall * Dd * sizeof(f16));
    f16*   vh     = (f16*)  carve((size_t)Bb * Sall * Dd * sizeof(f16));
    f16*   pet    = (f16*)  carve((size_t)Ll * HDc * sizeof(f16));
    float* attn_o = (float*)carve((size_t)Tt * Dd * sizeof(float));
    float* wo_o   = (float*)carve((size_t)Tt * Dd * sizeof(float));
    float* h1     = (float*)carve((size_t)Tt * Dd * sizeof(float));
    float* moe_o  = (float*)carve((size_t)Tt * Dd * sizeof(float));
    f16*   act    = (f16*)  carve((size_t)NASSIGN * INNERc * sizeof(f16));
    f16*   WqT    = (f16*)  carve((size_t)Dd * Dd * sizeof(f16));
    f16*   WkT    = (f16*)  carve((size_t)Dd * Dd * sizeof(f16));
    f16*   WvT    = (f16*)  carve((size_t)Dd * Dd * sizeof(f16));
    f16*   WoT    = (f16*)  carve((size_t)Dd * Dd * sizeof(f16));
    f16*   W1T    = (f16*)  carve((size_t)Ee * Dd * INNERc * sizeof(f16));
    f16*   W2T    = (f16*)  carve((size_t)Ee * INNERc * Dd * sizeof(f16));
    int*   topi   = (int*)  carve((size_t)NASSIGN * sizeof(int));
    float* topw   = (float*)carve((size_t)NASSIGN * sizeof(float));
    int*   rowmap = (int*)  carve((size_t)NASSIGN * sizeof(int));
    float* rowgw  = (float*)carve((size_t)NASSIGN * sizeof(float));
    int*   meta   = (int*)  carve(256);

    hipMemsetAsync(meta, 0, 256, stream);
    hipMemsetAsync(moe_o, 0, (size_t)Tt * Dd * sizeof(float), stream);

    dim3 blk256(256);

    // weight prep: f32 [K][N] -> f16 [N][K] (per matrix / per expert)
    k_w_t<<<dim3((Dd * Dd) / 256), blk256, 0, stream>>>(Wq, WqT, Dd, Dd);
    k_w_t<<<dim3((Dd * Dd) / 256), blk256, 0, stream>>>(Wk, WkT, Dd, Dd);
    k_w_t<<<dim3((Dd * Dd) / 256), blk256, 0, stream>>>(Wv, WvT, Dd, Dd);
    k_w_t<<<dim3((Dd * Dd) / 256), blk256, 0, stream>>>(Wo, WoT, Dd, Dd);
    k_w_t<<<dim3((Ee * Dd * INNERc) / 256), blk256, 0, stream>>>(W1, W1T, Dd, INNERc);
    k_w_t<<<dim3((Ee * INNERc * Dd) / 256), blk256, 0, stream>>>(W2, W2T, INNERc, Dd);

    // pe transpose
    k_pe_t<<<dim3(256), blk256, 0, stream>>>(key_pe, pet);

    // projections: q = h@Wq ; k/v over concat(h_cache, h)
    k_gemm<0, 0, false><<<dim3(4, 32, 1), blk256, 0, stream>>>(
        h, nullptr, nullptr, WqT, nullptr, nullptr, qh, nullptr, nullptr, meta, Tt, Dd, Dd);
    k_gemm<1, 0, false><<<dim3(4, 96, 1), blk256, 0, stream>>>(
        h, nullptr, hcache, WkT, nullptr, nullptr, kh, nullptr, nullptr, meta, Bb * Sall, Dd, Dd);
    k_gemm<1, 0, false><<<dim3(4, 96, 1), blk256, 0, stream>>>(
        h, nullptr, hcache, WvT, nullptr, nullptr, vh, nullptr, nullptr, meta, Bb * Sall, Dd, Dd);

    // banded flash attention (+ relative position term)
    k_attn<<<dim3(Bb * NH, Mm / 64), dim3(128), 0, stream>>>(qh, kh, vh, pet, attn_o);

    // output projection + residual LN
    k_gemm<0, 1, false><<<dim3(4, 32, 1), blk256, 0, stream>>>(
        attn_o, nullptr, nullptr, WoT, nullptr, wo_o, nullptr, nullptr, nullptr, meta, Tt, Dd, Dd);
    k_ln_resid<<<dim3(Tt), blk256, 0, stream>>>(h, wo_o, ln1g, ln1b, h1);

    // MoE routing: gate -> scan -> scatter
    k_gate<<<dim3(Tt / 8), blk256, 0, stream>>>(h1, Wg, bg, topi, topw, meta);
    k_scan<<<dim3(1), dim3(32), 0, stream>>>(meta);
    k_scatter<<<dim3(Tt / 256), blk256, 0, stream>>>(topi, topw, meta, rowmap, rowgw);

    // expert FFN (only routed tokens): act = relu(x@W1+b1) ; moe += w*(act@W2+b2)
    k_gemm<2, 2, true><<<dim3(INNERc / BNt, NASSIGN / BMt, Ee), blk256, 0, stream>>>(
        h1, nullptr, nullptr, W1T, b1, nullptr, act, rowmap, rowgw, meta, NASSIGN, INNERc, Dd);
    k_gemm<3, 3, false><<<dim3(Dd / BNt, NASSIGN / BMt, Ee), blk256, 0, stream>>>(
        nullptr, act, nullptr, W2T, b2, moe_o, nullptr, rowmap, rowgw, meta, NASSIGN, Dd, INNERc);

    // momentum + LN2 -> (h_out, mom_new)
    k_final<<<dim3(Tt), blk256, 0, stream>>>(h1, moe_o, momi, ln2g, ln2b, out);
}
